// RecursiveMamba2_v31_43250320670929
// MI455X (gfx1250) — matrix operand order repaired and checked by reading, not compiled, not run on hardware
//
#include <hip/hip_runtime.h>
#include <hip/hip_bf16.h>
#include <math.h>

// Problem constants (from reference)
#define NLOOPS  8
#define BATCH   8
#define LSEQ    2048
#define DIM     768
#define VOCAB   50288
#define HALT_IDC 50277
#define MROWS   (NLOOPS * BATCH)      // 64 rows of h
#define KSTEP   32                    // K per v_wmma_f32_16x16x32_bf16
#define NKS     (DIM / KSTEP)         // 24 wmma per 16x16 tile
#define STRIP_N 32                    // vocab rows staged per strip
#define NSTRIPS 8
#define NV_BLK  (STRIP_N * NSTRIPS)   // 256 vocab rows per block
#define NBLK2   ((VOCAB + NV_BLK - 1) / NV_BLK)  // 197
#define LDS_STRIDE 776                // 768 + 8 bf16 pad (bank-conflict break)

typedef __attribute__((ext_vector_type(16))) __bf16 v16bf;
typedef __attribute__((ext_vector_type(8)))  __bf16 v8bf;
typedef __attribute__((ext_vector_type(4)))  __bf16 v4bf;
typedef __attribute__((ext_vector_type(8)))  float  v8f;

// ---------------------------------------------------------------------------
// Phase 1: gather + RMSNorm -> h (bf16), and per-row target vocab ids
// ---------------------------------------------------------------------------
__global__ void __launch_bounds__(256)
k_prep(const float* __restrict__ x, const float* __restrict__ r,
       const float* __restrict__ nw, const int* __restrict__ ans_starts,
       const int* __restrict__ chain_targets, const int* __restrict__ chain_lens,
       __bf16* __restrict__ h_out, int* __restrict__ tgt_idx,
       float* __restrict__ tgt_logit)
{
    int m = blockIdx.x;               // m = l*BATCH + b
    int l = m / BATCH, b = m % BATCH;
    int t = threadIdx.x;
    int pos = ans_starts[b] - 1;
    pos = pos < 0 ? 0 : (pos > (LSEQ - 1) ? (LSEQ - 1) : pos);
    long long base = ((long long)m * LSEQ + pos) * DIM;

    float hv[3];
    float ss = 0.f;
#pragma unroll
    for (int i = 0; i < 3; ++i) {
        int d = t + i * 256;
        float v = x[base + d] + r[base + d];
        hv[i] = v;
        ss += v * v;
    }
    __shared__ float red[256];
    red[t] = ss;
    __syncthreads();
    for (int off = 128; off > 0; off >>= 1) {
        if (t < off) red[t] += red[t + off];
        __syncthreads();
    }
    float scale = rsqrtf(red[0] / (float)DIM + 1e-5f);
#pragma unroll
    for (int i = 0; i < 3; ++i) {
        int d = t + i * 256;
        h_out[(long long)m * DIM + d] = (__bf16)(hv[i] * scale * nw[d]);
    }
    if (t == 0) {
        int cl = chain_lens[b];
        int tix = (l < cl - 1) ? l : (cl - 1);   // min(l, cl-1)
        tgt_idx[m]   = chain_targets[b * NLOOPS + tix];
        tgt_logit[m] = 0.f;
    }
}

// ---------------------------------------------------------------------------
// Phase 2: streaming WMMA GEMV (64 x 768) * (768 x V) with fused online
// logsumexp / argmax / target-logit capture. 8 waves: 4 M-tiles x 2 N-halves.
// A fragments are strip-invariant; the compiler keeps all 24 resident in
// high VGPRs (s_set_vgpr_msb). B fragments are ping-pong double-buffered so
// each WMMA waits only on loads issued one step earlier.
// ---------------------------------------------------------------------------
__global__ void __launch_bounds__(256)
k_logits(const float* __restrict__ W, const __bf16* __restrict__ H,
         const int* __restrict__ tgt_idx, float* __restrict__ tgt_logit,
         float4* __restrict__ part)
{
    __shared__ __bf16 lds_w[STRIP_N * LDS_STRIDE];   // 49,664 B

    int t    = threadIdx.x;
    int blk  = blockIdx.x;
    int w    = t >> 5;        // wave 0..7
    int lane = t & 31;
    int l16  = lane & 15;
    int g    = lane >> 4;     // lane half (0/1)
    int mtile = w & 3;        // M-tile 0..3
    int half  = w >> 2;       // N-subtile within strip (0/1)

    int mg0 = mtile * 16 + 8 * g;    // first of this lane's 8 m-rows
    int tj[8];
#pragma unroll
    for (int j = 0; j < 8; ++j) tj[j] = tgt_idx[mg0 + j];

    float lsem[8], lses[8], bm[8];
    int   bi[8];
#pragma unroll
    for (int j = 0; j < 8; ++j) {
        lsem[j] = -__builtin_inff(); lses[j] = 0.f;
        bm[j]   = -__builtin_inff(); bi[j]   = VOCAB;
    }

    const __bf16* arow = H + (long long)(mtile * 16 + l16) * DIM;

    for (int s = 0; s < NSTRIPS; ++s) {
        int nbase = blk * NV_BLK + s * STRIP_N;

        // ---- stage W strip (32 rows x 768 f32), coalesced 16B loads -> bf16 LDS
        for (int i = t; i < STRIP_N * (DIM / 4); i += 256) {
            int rrow = i / (DIM / 4);
            int c4   = (i % (DIM / 4)) * 4;
            int rg   = nbase + rrow;
            float4 v = make_float4(0.f, 0.f, 0.f, 0.f);
            if (rg < VOCAB)
                v = *(const float4*)(W + (long long)rg * DIM + c4);
            v4bf pk = { (__bf16)v.x, (__bf16)v.y, (__bf16)v.z, (__bf16)v.w };
            *(v4bf*)&lds_w[rrow * LDS_STRIDE + c4] = pk;   // single b64 store
        }
        // hint-prefetch a slice of the next strip (global_prefetch_b8)
        if (s + 1 < NSTRIPS) {
            const char* p = (const char*)(W + (long long)(nbase + STRIP_N) * DIM);
            __builtin_prefetch(p + t * 384, 0, 1);
        }
        __syncthreads();

        // ---- 16x16 logits tile: 24 WMMAs over K=768, B ping-pong pipelined
        v8f c = {};
        const __bf16* brow = &lds_w[(half * 16 + l16) * LDS_STRIDE];

        v8bf b0lo = *(const v8bf*)(brow + 16 * g);
        v8bf b0hi = *(const v8bf*)(brow + 16 * g + 8);
#pragma unroll
        for (int k = 0; k < NKS; k += 2) {
            // issue B loads for step k+1 before consuming step k
            int k1 = (k + 1) * KSTEP;
            v8bf b1lo = *(const v8bf*)(brow + k1 + 16 * g);
            v8bf b1hi = *(const v8bf*)(brow + k1 + 16 * g + 8);

            v8bf a_lo = *(const v8bf*)(arow + k * KSTEP + 16 * g);
            v8bf a_hi = *(const v8bf*)(arow + k * KSTEP + 16 * g + 8);
            v16bf a0 = __builtin_shufflevector(a_lo, a_hi,
                        0,1,2,3,4,5,6,7,8,9,10,11,12,13,14,15);
            v16bf bb0 = __builtin_shufflevector(b0lo, b0hi,
                        0,1,2,3,4,5,6,7,8,9,10,11,12,13,14,15);
            c = __builtin_amdgcn_wmma_f32_16x16x32_bf16(
                    false, a0, false, bb0, (short)0, c, false, false);

            // issue B loads for step k+2 before consuming step k+1
            if (k + 2 < NKS) {
                int k2 = (k + 2) * KSTEP;
                b0lo = *(const v8bf*)(brow + k2 + 16 * g);
                b0hi = *(const v8bf*)(brow + k2 + 16 * g + 8);
            }
            v8bf a_lo1 = *(const v8bf*)(arow + k1 + 16 * g);
            v8bf a_hi1 = *(const v8bf*)(arow + k1 + 16 * g + 8);
            v16bf a1 = __builtin_shufflevector(a_lo1, a_hi1,
                        0,1,2,3,4,5,6,7,8,9,10,11,12,13,14,15);
            v16bf bb1 = __builtin_shufflevector(b1lo, b1hi,
                        0,1,2,3,4,5,6,7,8,9,10,11,12,13,14,15);
            c = __builtin_amdgcn_wmma_f32_16x16x32_bf16(
                    false, a1, false, bb1, (short)0, c, false, false);
        }

        // ---- fused epilogue: branchless online lse + argmax + target capture
        int n = nbase + half * 16 + l16;   // this lane's vocab column
        if (n < VOCAB) {
#pragma unroll
            for (int j = 0; j < 8; ++j) {
                float v  = c[j];
                float nm = fmaxf(lsem[j], v);
                lses[j]  = lses[j] * __expf(lsem[j] - nm) + __expf(v - nm);
                lsem[j]  = nm;
                bool gt  = v > bm[j];
                bm[j]    = gt ? v : bm[j];
                bi[j]    = gt ? n : bi[j];
                if (n == tj[j]) tgt_logit[mg0 + j] = v;   // unique writer
            }
        }
        __syncthreads();   // LDS reused next strip
    }

    // ---- cross-lane reduce over the 16 columns this wave-half covered
#pragma unroll
    for (int j = 0; j < 8; ++j) {
        float m1 = lsem[j], s1 = lses[j], v1 = bm[j];
        int   i1 = bi[j];
        for (int off = 1; off < 16; off <<= 1) {
            float m2 = __shfl_xor(m1, off);
            float s2 = __shfl_xor(s1, off);
            float v2 = __shfl_xor(v1, off);
            int   i2 = __shfl_xor(i1, off);
            if (s2 > 0.f) {
                if (s1 <= 0.f)      { m1 = m2; s1 = s2; }
                else if (m2 > m1)   { s1 = s2 + s1 * __expf(m1 - m2); m1 = m2; }
                else                { s1 = s1 + s2 * __expf(m2 - m1); }
            }
            if (v2 > v1 || (v2 == v1 && i2 < i1)) { v1 = v2; i1 = i2; }
        }
        lsem[j] = m1; lses[j] = s1; bm[j] = v1; bi[j] = i1;
    }
    if (l16 == 0) {
#pragma unroll
        for (int j = 0; j < 8; ++j)
            part[(blk * 2 + half) * MROWS + mg0 + j] =
                make_float4(lsem[j], lses[j], bm[j], (float)bi[j]);
    }
}

// ---------------------------------------------------------------------------
// Phase 3: reduce partials per row, then the reference's scalar epilogue.
// ---------------------------------------------------------------------------
__global__ void __launch_bounds__(64)
k_final(const float4* __restrict__ part, const float* __restrict__ tgt_logit,
        const int* __restrict__ tgt_idx, const int* __restrict__ ans_starts,
        float* __restrict__ out)
{
    __shared__ float s_ce[MROWS], s_co[MROWS];
    __shared__ int   s_tg[MROWS];
    int t = threadIdx.x;
    if (t < MROWS) {
        float m1 = -__builtin_inff(), s1 = 0.f, v1 = -__builtin_inff();
        int   i1 = VOCAB;
        for (int p = 0; p < NBLK2 * 2; ++p) {
            float4 rec = part[p * MROWS + t];
            float m2 = rec.x, s2 = rec.y, v2 = rec.z;
            int   i2 = (int)rec.w;
            if (s2 > 0.f) {
                if (s1 <= 0.f)    { m1 = m2; s1 = s2; }
                else if (m2 > m1) { s1 = s2 + s1 * __expf(m1 - m2); m1 = m2; }
                else              { s1 = s1 + s2 * __expf(m2 - m1); }
            }
            if (v2 > v1 || (v2 == v1 && i2 < i1)) { v1 = v2; i1 = i2; }
        }
        float logZ = m1 + __logf(s1);
        s_ce[t] = logZ - tgt_logit[t];
        s_co[t] = (i1 == tgt_idx[t]) ? 1.f : 0.f;
        s_tg[t] = tgt_idx[t];
    }
    __syncthreads();
    if (t == 0) {
        float loop_loss[NLOOPS], loop_acc[NLOOPS];
        int   has[NLOOPS];
        float n_has = 0.f;
        for (int l = 0; l < NLOOPS; ++l) {
            float cnt = 0.f, sl = 0.f, sa = 0.f;
            for (int b = 0; b < BATCH; ++b) {
                int as = ans_starts[b];
                int m = l * BATCH + b;
                float vf = (as >= 1 && as < LSEQ && s_tg[m] < VOCAB) ? 1.f : 0.f;
                cnt += vf; sl += s_ce[m] * vf; sa += s_co[m] * vf;
            }
            has[l] = (cnt > 0.f);
            float denom = cnt > 1.f ? cnt : 1.f;
            loop_loss[l] = has[l] ? sl / denom : 0.f;
            loop_acc[l]  = has[l] ? sa / denom : 0.f;
            n_has += has[l] ? 1.f : 0.f;
        }
        float avg_loss = 0.f, avg_acc = 0.f;
        for (int l = 0; l < NLOOPS; ++l)
            if (has[l]) { avg_loss += loop_loss[l]; avg_acc += loop_acc[l]; }
        float dn = n_has > 1.f ? n_has : 1.f;
        avg_loss /= dn; avg_acc /= dn;

        int last_valid = NLOOPS - 1;          // all-false reversed argmax -> 0
        for (int l = NLOOPS - 1; l >= 0; --l)
            if (has[l]) { last_valid = l; break; }

        float mask_sum = 0.f, ans_sum = 0.f;
        for (int l = 0; l < NLOOPS; ++l) {
            float msk = has[l] ? 1.f : 0.f;
            if (n_has > 1.f && l == last_valid) msk = 0.f;
            mask_sum += msk; ans_sum += loop_acc[l] * msk;
        }
        float answer_acc = (mask_sum > 0.f)
            ? ans_sum / (mask_sum > 1.f ? mask_sum : 1.f) : avg_acc;

        float hm_sum = 0.f, hc_sum = 0.f;
        for (int l = 0; l < NLOOPS; ++l)
            for (int b = 0; b < BATCH; ++b) {
                int as = ans_starts[b];
                int m = l * BATCH + b;
                float vf = (as >= 1 && as < LSEQ && s_tg[m] < VOCAB) ? 1.f : 0.f;
                float hmf = vf * ((s_tg[m] == HALT_IDC) ? 1.f : 0.f);
                hm_sum += hmf; hc_sum += s_co[m] * hmf;
            }
        float halt_acc = (hm_sum > 0.f)
            ? hc_sum / (hm_sum > 1.f ? hm_sum : 1.f) : 0.f;

        out[0] = avg_loss; out[1] = avg_acc; out[2] = answer_acc; out[3] = halt_acc;
    }
}

// ---------------------------------------------------------------------------
extern "C" void kernel_launch(void* const* d_in, const int* in_sizes, int n_in,
                              void* d_out, int out_size, void* d_ws, size_t ws_size,
                              hipStream_t stream)
{
    (void)in_sizes; (void)n_in; (void)out_size; (void)ws_size;
    const float* x  = (const float*)d_in[0];   // x_steps
    const float* r  = (const float*)d_in[1];   // residual_steps
    const float* nw = (const float*)d_in[2];   // norm_weight
    const float* W  = (const float*)d_in[3];   // lm_head_weight (V x D)
    const int*   as = (const int*)d_in[4];     // ans_starts
    const int*   ct = (const int*)d_in[5];     // chain_targets
    const int*   cl = (const int*)d_in[6];     // chain_lens

    char* ws = (char*)d_ws;
    __bf16* h        = (__bf16*)ws;                 // 64*768*2   = 98304 B
    int*    tgt_idx  = (int*)   (ws + 98304);       // 64*4       = 256 B
    float*  tgt_log  = (float*) (ws + 98560);       // 64*4       = 256 B
    float4* part     = (float4*)(ws + 98816);       // 394*64*16  = 403456 B

    k_prep  <<<MROWS, 256, 0, stream>>>(x, r, nw, as, ct, cl, h, tgt_idx, tgt_log);
    k_logits<<<NBLK2, 256, 0, stream>>>(W, h, tgt_idx, tgt_log, part);
    k_final <<<1, 64, 0, stream>>>(part, tgt_log, tgt_idx, as, (float*)d_out);
}
